// DNA_Module_10780367912992
// MI455X (gfx1250) — compile-verified, matrix operand order checked by reading, not run on hardware
//
#include <hip/hip_runtime.h>

// ---------------------------------------------------------------------------
// Types for CDNA5 WMMA (wave32)
// ---------------------------------------------------------------------------
typedef __bf16 bf16_t;
typedef __attribute__((ext_vector_type(16))) __bf16 v16bf;
typedef __attribute__((ext_vector_type(8)))  float  v8f;

// exact parameter types of __builtin_amdgcn_global_load_async_to_lds_b128
// (from hipcc diagnostics): 16-byte int vectors in global / shared space
typedef __attribute__((__vector_size__(4 * sizeof(int)))) int gvec4i;
typedef __attribute__((address_space(1))) gvec4i as1_v4i;
typedef __attribute__((address_space(3))) gvec4i as3_v4i;

#define NTOK 4096   // 64*64 tokens
#define EMB  256

// CDNA5 async global->LDS path (gate on builtin availability so the file
// always compiles; device pass takes the async branch on gfx1250).
#if defined(__AMDGCN__) && \
    __has_builtin(__builtin_amdgcn_global_load_async_to_lds_b128) && \
    __has_builtin(__builtin_amdgcn_s_wait_asynccnt)
#define USE_ASYNC_LDS 1
#else
#define USE_ASYNC_LDS 0
#endif

// ---------------------------------------------------------------------------
// f32 -> bf16 convert (elementwise)
// ---------------------------------------------------------------------------
__global__ void cvt_f32_bf16(const float* __restrict__ s, bf16_t* __restrict__ d, int n) {
  int i = blockIdx.x * blockDim.x + threadIdx.x;
  if (i < n) d[i] = (bf16_t)s[i];
}

// cnn_embeddings [256, 4096] (ch-major) -> [4096, 256] bf16 (token-major)
__global__ void transpose_cvt(const float* __restrict__ s, bf16_t* __restrict__ d) {
  int i = blockIdx.x * blockDim.x + threadIdx.x;
  if (i < NTOK * EMB) {
    int n = i >> 8, c = i & 255;
    d[i] = (bf16_t)s[c * NTOK + n];
  }
}

// ---------------------------------------------------------------------------
// WMMA fragment load: 16x32 bf16 tile from row-major [rows, ld] at (row0,k0).
// ISA 16-bit A layout: lanes 0-15 -> M=lane, K k0..k0+7 (elems 0-7) and
// k0+16..k0+23 (elems 8-15); lanes 16-31 -> same M, K +8 / +24.
// B operand uses the identical pattern on W's rows (W is [N,K] = B^T rows).
// ---------------------------------------------------------------------------
__device__ inline v16bf frag_ld(const bf16_t* __restrict__ base, int ld,
                                int row0, int k0, int lane) {
  const bf16_t* p = base + (size_t)(row0 + (lane & 15)) * ld + (k0 + ((lane >> 4) << 3));
  v16bf f;
  ((uint4*)&f)[0] = *(const uint4*)p;          // K k0..k0+7   (or +8)
  ((uint4*)&f)[1] = *(const uint4*)(p + 16);   // K k0+16..+23 (or +24)
  return f;
}

// ---------------------------------------------------------------------------
// Tiled bf16 WMMA GEMM:  C[M,N] = A[M,K] * W[N,K]^T + bias
// Block: 256 threads = 8 waves; block tile 128x64 (wave tile 16x64, 4 wmma).
// B tile (64x32 bf16, 4KB) double-buffered in LDS; staged with CDNA5
// GLOBAL_LOAD_ASYNC_TO_LDS (ASYNCcnt) so the next K-tile streams into LDS
// while the WMMAs consume the current one.
// epi: 0 = store f32 to Cf ; 1 = store bf16 to Cb ; 2 = exact GELU + bf16.
// Requires M%128==0, N%64==0, K%32==0.
// ---------------------------------------------------------------------------
__global__ __launch_bounds__(256) void gemm_bf16(
    const bf16_t* __restrict__ A, const bf16_t* __restrict__ Wt,
    const float* __restrict__ bias, float* __restrict__ Cf,
    bf16_t* __restrict__ Cb, int M, int N, int K, int ldc, int colOff, int epi)
{
  __shared__ __align__(16) bf16_t Bs[2][64 * 32];
  const int tid  = threadIdx.x;
  const int lane = tid & 31;
  const int wave = tid >> 5;
  const int row0 = blockIdx.x * 128 + wave * 16;
  const int col0 = blockIdx.y * 64;

  // this thread's 16B slice of the 64x32 B tile
  const int r  = tid >> 2;
  const int ck = (tid & 3) << 3;
  const bf16_t* wsrc = Wt + (size_t)(col0 + r) * K + ck;   // +k0 per step

  v8f acc[4];
  #pragma unroll
  for (int t = 0; t < 4; ++t)
    #pragma unroll
    for (int v = 0; v < 8; ++v) acc[t][v] = 0.0f;

#if USE_ASYNC_LDS
  // prologue: stage k=0 into buffer 0 (16B per thread, straight to LDS)
  __builtin_amdgcn_global_load_async_to_lds_b128(
      (as1_v4i*)(wsrc),
      (as3_v4i*)&Bs[0][r * 32 + ck], 0, 0);
  int buf = 0;
  for (int k0 = 0; k0 < K; k0 += 32, buf ^= 1) {
    if (k0 + 32 < K) {
      __builtin_amdgcn_global_load_async_to_lds_b128(
          (as1_v4i*)(wsrc + k0 + 32),
          (as3_v4i*)&Bs[buf ^ 1][r * 32 + ck], 0, 0);
      __builtin_amdgcn_s_wait_asynccnt(1);   // batch for Bs[buf] has landed
    } else {
      __builtin_amdgcn_s_wait_asynccnt(0);
    }
    __syncthreads();                          // make it block-visible

    v16bf a = frag_ld(A, K, row0, k0, lane);
    #pragma unroll
    for (int t = 0; t < 4; ++t) {
      const bf16_t* p = &Bs[buf][(t * 16 + (lane & 15)) * 32 + ((lane >> 4) << 3)];
      v16bf b;
      ((uint4*)&b)[0] = *(const uint4*)p;
      ((uint4*)&b)[1] = *(const uint4*)(p + 16);
      acc[t] = __builtin_amdgcn_wmma_f32_16x16x32_bf16(
          false, a, false, b, (short)0, acc[t], false, false);
    }
    __syncthreads();   // all reads of Bs[buf] done before it is re-staged
  }
#else
  // fallback: synchronous staging (single buffer)
  for (int k0 = 0; k0 < K; k0 += 32) {
    *(uint4*)&Bs[0][r * 32 + ck] = *(const uint4*)(wsrc + k0);
    __syncthreads();
    v16bf a = frag_ld(A, K, row0, k0, lane);
    #pragma unroll
    for (int t = 0; t < 4; ++t) {
      const bf16_t* p = &Bs[0][(t * 16 + (lane & 15)) * 32 + ((lane >> 4) << 3)];
      v16bf b;
      ((uint4*)&b)[0] = *(const uint4*)p;
      ((uint4*)&b)[1] = *(const uint4*)(p + 16);
      acc[t] = __builtin_amdgcn_wmma_f32_16x16x32_bf16(
          false, a, false, b, (short)0, acc[t], false, false);
    }
    __syncthreads();
  }
#endif

  // C/D layout: VGPR v, lane L -> row = v + 8*(L>>4), col = L&15
  const int rb = row0 + ((lane >> 4) << 3);
  const int nl = lane & 15;
  #pragma unroll
  for (int t = 0; t < 4; ++t) {
    int col = col0 + t * 16 + nl;
    float bv = bias ? bias[col] : 0.0f;
    #pragma unroll
    for (int v = 0; v < 8; ++v) {
      float val = acc[t][v] + bv;
      if (epi == 2) val = 0.5f * val * (1.0f + erff(val * 0.70710678118654752f));
      if (epi == 0) Cf[(size_t)(rb + v) * ldc + col] = val;
      else          Cb[(size_t)(rb + v) * ldc + colOff + col] = (bf16_t)val;
    }
  }
}

// ---------------------------------------------------------------------------
// Neighborhood attention, sigmoid gating, with the reference's unfold
// scramble: element (p,c) maps to neighbor q=(p*32+c)%49, channel d=(p*32+c)/49.
// One wave per (pixel, head); lane = channel c.
// ---------------------------------------------------------------------------
__global__ __launch_bounds__(256) void na_attn(
    const float* __restrict__ Q, const float* __restrict__ Km,
    const float* __restrict__ Vm, const float* __restrict__ rpb,
    bf16_t* __restrict__ out)
{
  const int n = blockIdx.x;
  const int h = threadIdx.x >> 5;
  const int c = threadIdx.x & 31;
  const int y = n >> 6, x = n & 63;
  const float q = Q[(size_t)n * EMB + h * 32 + c];
  float acc = 0.0f;

  #pragma unroll 1
  for (int p = 0; p < 49; ++p) {
    int idx = p * 32 + c;           // 0..1567
    int qq  = idx % 49;             // scrambled neighbor index
    int d   = idx / 49;             // scrambled channel
    int i = qq / 7, j = qq % 7;
    int sy = y - 9 + 3 * i;         // pad = dil*(KS-1)/2 = 9
    int sx = x - 9 + 3 * j;
    bool ok = (sy >= 0) && (sy < 64) && (sx >= 0) && (sx < 64);
    int m = ok ? (sy * 64 + sx) : 0;
    size_t o = (size_t)m * EMB + h * 32 + d;
    float kv = ok ? Km[o] : 0.0f;
    float vv = ok ? Vm[o] : 0.0f;

    float part = q * kv;
    #pragma unroll
    for (int off = 16; off > 0; off >>= 1) part += __shfl_xor(part, off, 32);

    float s  = part * 0.17677669529663687f + rpb[h * 49 + p];  // 1/sqrt(32)
    float pr = 1.0f / (1.0f + __expf(-s));
    acc += pr * vv;
  }
  out[(size_t)n * EMB + h * 32 + c] = (bf16_t)acc;
}

// ---------------------------------------------------------------------------
// Block-wide (256-thread) sum
// ---------------------------------------------------------------------------
__device__ inline float block_sum(float v) {
  __shared__ float s[9];
  #pragma unroll
  for (int off = 16; off > 0; off >>= 1) v += __shfl_xor(v, off, 32);
  __syncthreads();
  if ((threadIdx.x & 31) == 0) s[threadIdx.x >> 5] = v;
  __syncthreads();
  if (threadIdx.x == 0) { float t = 0; for (int i = 0; i < 8; ++i) t += s[i]; s[8] = t; }
  __syncthreads();
  return s[8];
}

// sam = LN(mlp_out, ln_sam); out = LN(img + sam, norm_sam)      [row-major]
__global__ __launch_bounds__(256) void final_sam(
    const float* __restrict__ m2, const float* __restrict__ img,
    const float* __restrict__ g1, const float* __restrict__ b1,
    const float* __restrict__ g2, const float* __restrict__ b2,
    float* __restrict__ out)
{
  int n = blockIdx.x, c = threadIdx.x;
  float x  = m2[(size_t)n * EMB + c];
  float mu = block_sum(x) * (1.0f / EMB);
  float va = block_sum((x - mu) * (x - mu)) * (1.0f / EMB);
  float sam = (x - mu) * rsqrtf(va + 1e-5f) * g1[c] + b1[c];
  float y  = img[(size_t)n * EMB + c] + sam;
  float m2_ = block_sum(y) * (1.0f / EMB);
  float v2  = block_sum((y - m2_) * (y - m2_)) * (1.0f / EMB);
  out[(size_t)n * EMB + c] = (y - m2_) * rsqrtf(v2 + 1e-5f) * g2[c] + b2[c];
}

// cnf = LN(mlp_out, ln_cnn); res[c] = cnn[c,n] + cnf[c]; out[c,n] = LN_c(res)
__global__ __launch_bounds__(256) void final_cnn(
    const float* __restrict__ m2, const float* __restrict__ cnn,
    const float* __restrict__ g1, const float* __restrict__ b1,
    const float* __restrict__ g2, const float* __restrict__ b2,
    float* __restrict__ out)
{
  int n = blockIdx.x, c = threadIdx.x;
  float x  = m2[(size_t)n * EMB + c];
  float mu = block_sum(x) * (1.0f / EMB);
  float va = block_sum((x - mu) * (x - mu)) * (1.0f / EMB);
  float cnf = (x - mu) * rsqrtf(va + 1e-5f) * g1[c] + b1[c];
  float y  = cnn[(size_t)c * NTOK + n] + cnf;
  float m2_ = block_sum(y) * (1.0f / EMB);
  float v2  = block_sum((y - m2_) * (y - m2_)) * (1.0f / EMB);
  out[(size_t)c * NTOK + n] = (y - m2_) * rsqrtf(v2 + 1e-5f) * g2[c] + b2[c];
}

// ---------------------------------------------------------------------------
// Host orchestration
// ---------------------------------------------------------------------------
extern "C" void kernel_launch(void* const* d_in, const int* in_sizes, int n_in,
                              void* d_out, int out_size, void* d_ws, size_t ws_size,
                              hipStream_t stream)
{
  (void)n_in; (void)out_size; (void)ws_size;
  const float* img = (const float*)d_in[0];   // [1,64,64,256]
  const float* cnn = (const float*)d_in[1];   // [1,256,64,64]
  const float* rpb = (const float*)d_in[2];   // [8,7,7]

  char* base = (char*)d_ws;
  size_t off = 0;
  auto take = [&](size_t bytes) -> void* {
    void* p = base + off;
    off += bytes; off = (off + 255) & ~(size_t)255;
    return p;
  };

  bf16_t* img_bf = (bf16_t*)take((size_t)NTOK * EMB * 2);
  bf16_t* cnn_bf = (bf16_t*)take((size_t)NTOK * EMB * 2);

  // weight indices (dict insertion order of setup_inputs)
  static const int widx[22] = { 3,5,7,9,11,13,15,17,19,21,23,25,   // 12 qkv .w
                                27,29,31,33,                        // 4 out .w
                                35,37,                              // fuse .w
                                39,41,43,45 };                      // mlp .w
  bf16_t* wbf[64] = {};
  for (int i = 0; i < 22; ++i)
    wbf[widx[i]] = (bf16_t*)take((size_t)in_sizes[widx[i]] * 2);

  float*  Qf      = (float*)take((size_t)NTOK * EMB * 4);
  float*  Kf      = (float*)take((size_t)NTOK * EMB * 4);
  float*  Vf      = (float*)take((size_t)NTOK * EMB * 4);
  bf16_t* attn_bf = (bf16_t*)take((size_t)NTOK * EMB * 2);
  bf16_t* sam_cat = (bf16_t*)take((size_t)NTOK * 512 * 2);
  bf16_t* cnn_cat = (bf16_t*)take((size_t)NTOK * 512 * 2);
  bf16_t* fuse_bf = (bf16_t*)take((size_t)NTOK * EMB * 2);
  bf16_t* h1_bf   = (bf16_t*)take((size_t)NTOK * 1024 * 2);
  float*  m2_f    = (float*)take((size_t)NTOK * EMB * 4);

  // --- precision conversion ---
  cvt_f32_bf16<<<(NTOK * EMB + 255) / 256, 256, 0, stream>>>(img, img_bf, NTOK * EMB);
  transpose_cvt<<<(NTOK * EMB + 255) / 256, 256, 0, stream>>>(cnn, cnn_bf);
  for (int i = 0; i < 22; ++i) {
    int id = widx[i], n = in_sizes[id];
    cvt_f32_bf16<<<(n + 255) / 256, 256, 0, stream>>>((const float*)d_in[id], wbf[id], n);
  }

  auto gemm = [&](const bf16_t* A, int id_w, int id_b, float* Cf, bf16_t* Cb,
                  int M, int N, int K, int ldc, int colOff, int epi) {
    dim3 g(M / 128, N / 64);
    gemm_bf16<<<g, 256, 0, stream>>>(A, wbf[id_w], (const float*)d_in[id_b],
                                     Cf, Cb, M, N, K, ldc, colOff, epi);
  };

  // --- 4 attention branches: QKV -> NA -> out-proj into concat buffer ---
  struct Br { const bf16_t* xq; const bf16_t* xkv; int q, k, v, o; bf16_t* cat; int co; };
  const Br brs[4] = {
    { img_bf, img_bf,  3,  5,  7, 27, sam_cat,   0 },   // ss
    { img_bf, cnn_bf,  9, 11, 13, 29, sam_cat, 256 },   // sc
    { cnn_bf, cnn_bf, 15, 17, 19, 31, cnn_cat,   0 },   // cc
    { cnn_bf, img_bf, 21, 23, 25, 33, cnn_cat, 256 },   // cs
  };
  for (int b = 0; b < 4; ++b) {
    const Br& r = brs[b];
    gemm(r.xq,  r.q, r.q + 1, Qf, nullptr, NTOK, 256, 256, 256, 0, 0);
    gemm(r.xkv, r.k, r.k + 1, Kf, nullptr, NTOK, 256, 256, 256, 0, 0);
    gemm(r.xkv, r.v, r.v + 1, Vf, nullptr, NTOK, 256, 256, 256, 0, 0);
    na_attn<<<NTOK, 256, 0, stream>>>(Qf, Kf, Vf, rpb, attn_bf);
    gemm(attn_bf, r.o, r.o + 1, nullptr, r.cat, NTOK, 256, 256, 512, r.co, 1);
  }

  float* out_sam = (float*)d_out;
  float* out_cnn = (float*)d_out + (size_t)NTOK * EMB;

  // --- SAM pipeline: fuse -> MLP(gelu) -> LN chain ---
  gemm(sam_cat, 35, 36, nullptr, fuse_bf, NTOK,  256,  512,  256, 0, 1);
  gemm(fuse_bf, 39, 40, nullptr, h1_bf,   NTOK, 1024,  256, 1024, 0, 2);
  gemm(h1_bf,   41, 42, m2_f,    nullptr, NTOK,  256, 1024,  256, 0, 0);
  final_sam<<<NTOK, 256, 0, stream>>>(m2_f, img,
      (const float*)d_in[47], (const float*)d_in[48],
      (const float*)d_in[51], (const float*)d_in[52], out_sam);

  // --- CNN pipeline (reuses fuse_bf / h1_bf / m2_f; stream-ordered) ---
  gemm(cnn_cat, 37, 38, nullptr, fuse_bf, NTOK,  256,  512,  256, 0, 1);
  gemm(fuse_bf, 43, 44, nullptr, h1_bf,   NTOK, 1024,  256, 1024, 0, 2);
  gemm(h1_bf,   45, 46, m2_f,    nullptr, NTOK,  256, 1024,  256, 0, 0);
  final_cnn<<<NTOK, 256, 0, stream>>>(m2_f, cnn,
      (const float*)d_in[49], (const float*)d_in[50],
      (const float*)d_in[53], (const float*)d_in[54], out_cnn);
}
